// MHA_53592601919963
// MI455X (gfx1250) — matrix operand order repaired and checked by reading, not compiled, not run on hardware
//
#include <hip/hip_runtime.h>
#include <hip/hip_bf16.h>

// ---------------------------------------------------------------------------
// Local-window causal MHA for MI455X (gfx1250).
//  - All four matmul stages on v_wmma_f32_16x16x32_f16 (f16 in, f32 acc).
//  - GEMM kernels: 32x64 tile per wave (8 accumulators) for A/B frag reuse.
//  - Attention: flash-style online softmax over 32-key blocks; K/V tiles
//    staged into LDS with double-buffered GLOBAL_LOAD_ASYNC_TO_LDS_B128
//    (ASYNCcnt), P transposed C-layout -> A-layout through LDS (DScnt).
// ---------------------------------------------------------------------------

typedef __attribute__((ext_vector_type(16))) _Float16 v16h;
typedef __attribute__((ext_vector_type(8)))  float    v8f;

union F16x16 { v16h v; _Float16 h[16]; uint4 q[2]; };

static constexpr int    Bc   = 4;
static constexpr int    Tc   = 2048;
static constexpr int    Dc   = 512;     // model dim
static constexpr int    Hc   = 8;       // heads
static constexpr int    HD   = 64;      // head dim
static constexpr int    DQc  = 512;     // H * HD
static constexpr int    WLOC = 256;     // local_attn_size
#define SCORE_SCALE 0.02209708691207961f   /* 1/sqrt(T=2048) */
#define NEGBIG (-1.0e30f)

static __device__ __forceinline__ v8f wmma16(const F16x16& a, const F16x16& b, v8f c) {
  // D = A(16x32 f16) * B(32x16 f16) + C(16x16 f32)
  return __builtin_amdgcn_wmma_f32_16x16x32_f16(false, a.v, false, b.v,
                                                (short)0, c, false, false);
}

// ---------------------------------------------------------------------------
// Kernel 1: fused QKV projection.  z = 0/1/2 selects (q_in,Wq,bq)/(k,..)/(v,..)
// One wave computes a 32x64 tile of Y = X @ W + b (8 WMMAs per k-step;
// A frags reused across 4 n-tiles, B frags across 2 row-tiles).
// K output stored transposed per head: [b][h][d][t].
// ---------------------------------------------------------------------------
__global__ __launch_bounds__(128) void k_qkv(
    const float* __restrict__ qin, const float* __restrict__ kin,
    const float* __restrict__ vin,
    const float* __restrict__ Wq, const float* __restrict__ bq,
    const float* __restrict__ Wk, const float* __restrict__ bk,
    const float* __restrict__ Wv, const float* __restrict__ bv,
    _Float16* __restrict__ Qh, _Float16* __restrict__ Kt,
    _Float16* __restrict__ Vh)
{
  const int lane  = threadIdx.x;                 // 0..31
  const int which = blockIdx.z;                  // 0=q 1=k 2=v
  const float* X    = (which == 0) ? qin : (which == 1) ? kin : vin;
  const float* W    = (which == 0) ? Wq  : (which == 1) ? Wk  : Wv;
  const float* bias = (which == 0) ? bq  : (which == 1) ? bk  : bv;

  const int row0 = blockIdx.x * 32;              // global row in [0, B*T)
  const int n0   = (blockIdx.y * blockDim.y + threadIdx.y) * 64;
  const int r    = lane & 15;
  const int hi0  = (lane < 16) ? 0 : 8;          // A k-base AND C row-offset

  v8f acc[2][4] = {};
  for (int kb = 0; kb < Dc; kb += 32) {
    // A fragments: two 16x32 tiles of X (rows row0.. / row0+16..), f32 -> f16.
    F16x16 a[2];
#pragma unroll
    for (int rt = 0; rt < 2; ++rt) {
      const float* xr = X + (size_t)(row0 + rt * 16 + r) * Dc + kb;
#pragma unroll
      for (int run = 0; run < 2; ++run) {
        const float4* x4 = (const float4*)(xr + hi0 + run * 16);
        float4 u0 = x4[0], u1 = x4[1];
        a[rt].h[run * 8 + 0] = (_Float16)u0.x; a[rt].h[run * 8 + 1] = (_Float16)u0.y;
        a[rt].h[run * 8 + 2] = (_Float16)u0.z; a[rt].h[run * 8 + 3] = (_Float16)u0.w;
        a[rt].h[run * 8 + 4] = (_Float16)u1.x; a[rt].h[run * 8 + 5] = (_Float16)u1.y;
        a[rt].h[run * 8 + 6] = (_Float16)u1.z; a[rt].h[run * 8 + 7] = (_Float16)u1.w;
      }
    }
    // B fragments: four 32x16 tiles of W.  lane -> n col, elements -> k (table order).
#pragma unroll
    for (int ct = 0; ct < 4; ++ct) {
      F16x16 bf;
      const float* wc = W + (size_t)kb * DQc + n0 + ct * 16 + r;
#pragma unroll
      for (int s = 0; s < 16; ++s)
        bf.h[s] = (_Float16)wc[(size_t)(hi0 + (s & 7) + ((s >> 3) << 4)) * DQc];
#pragma unroll
      for (int rt = 0; rt < 2; ++rt)
        acc[rt][ct] = wmma16(a[rt], bf, acc[rt][ct]);
    }
  }

  // Epilogue: bias + store f16.  C layout: vgpr j -> row (j + hi0), lane -> col.
  const int bb = row0 >> 11;                     // row0 / T
  const int tbase = row0 & (Tc - 1);
#pragma unroll
  for (int ct = 0; ct < 4; ++ct) {
    const int col = n0 + ct * 16 + r;
    const float bcol = bias[col];
    const int hh = col >> 6, dd = col & 63;
#pragma unroll
    for (int rt = 0; rt < 2; ++rt)
#pragma unroll
      for (int j = 0; j < 8; ++j) {
        const int t = tbase + rt * 16 + j + hi0;
        const float y = acc[rt][ct][j] + bcol;
        if (which == 1) {
          Kt[(((size_t)bb * Hc + hh) * HD + dd) * Tc + t] = (_Float16)y; // [b][h][d][t]
        } else {
          _Float16* P = (which == 0) ? Qh : Vh;
          P[(((size_t)bb * Hc + hh) * Tc + t) * HD + dd] = (_Float16)y;  // [b][h][t][d]
        }
      }
  }
}

// ---------------------------------------------------------------------------
// Async staging of one 32-key block: K-region (64 d-rows x 32 keys, 4 KB) and
// V-region (32 keys x 64 vd, 4 KB) via GLOBAL_LOAD_ASYNC_TO_LDS_B128.
// 16 async instructions per call (8 K + 8 V); tracked on ASYNCcnt.
// ---------------------------------------------------------------------------
static __device__ __forceinline__ void issue_async_kv(
    const _Float16* __restrict__ gK,   // Kt base for (b,h): rows of T halves
    const _Float16* __restrict__ gV,   // V  base for (b,h): rows of 64 halves
    int kb, unsigned ldsK, unsigned ldsV, int lane)
{
  // Drain the DS pipe first: ASYNC writes are unordered vs in-flight ds reads
  // of the buffer being overwritten.
  asm volatile("s_wait_dscnt 0" ::: "memory");
#pragma unroll
  for (int it = 0; it < 8; ++it) {
    const int c0 = it * 32 + lane;
    // K: chunk c0 -> d-row (c0>>2), 16B sub-chunk (c0&3).
    {
      const int d = c0 >> 2, qk = c0 & 3;
      unsigned gofs = (unsigned)((d * Tc + kb) * 2 + qk * 16);
      unsigned lofs = ldsK + (unsigned)(d * 64 + qk * 16);
      asm volatile("global_load_async_to_lds_b128 %0, %1, %2"
                   :: "v"(lofs), "v"(gofs), "s"(gK) : "memory");
    }
    // V: chunk c0 -> key row (c0>>3), 16B sub-chunk (c0&7).
    {
      const int row = c0 >> 3, qv = c0 & 7;
      unsigned gofs = (unsigned)(((kb + row) * HD) * 2 + qv * 16);
      unsigned lofs = ldsV + (unsigned)(row * 128 + qv * 16);
      asm volatile("global_load_async_to_lds_b128 %0, %1, %2"
                   :: "v"(lofs), "v"(gofs), "s"(gV) : "memory");
    }
  }
}

// ---------------------------------------------------------------------------
// Kernel 2: banded causal attention, one wave per (b, h, 16-query tile).
// Double-buffered async K/V staging; online softmax; P via LDS transpose.
// ---------------------------------------------------------------------------
__global__ __launch_bounds__(32) void k_attn(
    const _Float16* __restrict__ Qh, const _Float16* __restrict__ KtB,
    const _Float16* __restrict__ Vh, _Float16* __restrict__ attn)
{
  __shared__ _Float16 Kl[2][64 * 32];   // [buf][d][key]   4 KB each
  __shared__ _Float16 Vl[2][32 * 64];   // [buf][key][vd]  4 KB each
  __shared__ _Float16 pl[16 * 32];      // P transpose tile

  const int lane = threadIdx.x;
  const int q0 = blockIdx.x * 16;
  const int h  = blockIdx.y;
  const int b  = blockIdx.z;
  const _Float16* Q  = Qh  + (((size_t)b * Hc + h) * Tc) * HD;
  const _Float16* Kt = KtB + (((size_t)b * Hc + h) * HD) * Tc;
  const _Float16* V  = Vh  + (((size_t)b * Hc + h) * Tc) * HD;

  const unsigned ldsK[2] = { (unsigned)(size_t)&Kl[0][0], (unsigned)(size_t)&Kl[1][0] };
  const unsigned ldsV[2] = { (unsigned)(size_t)&Vl[0][0], (unsigned)(size_t)&Vl[1][0] };

  const int r   = lane & 15;
  const int hi0 = (lane < 16) ? 0 : 8;

  // Persistent Q A-fragments (d = 0..31, 32..63).
  F16x16 qa[2];
#pragma unroll
  for (int c = 0; c < 2; ++c) {
    const _Float16* qr = Q + (size_t)(q0 + r) * HD + c * 32 + hi0;
    qa[c].q[0] = *(const uint4*)(qr);
    qa[c].q[1] = *(const uint4*)(qr + 16);
  }

  float mrun[8], lrun[8];
#pragma unroll
  for (int j = 0; j < 8; ++j) { mrun[j] = NEGBIG; lrun[j] = 0.0f; }
  v8f o[4] = {};

  int kb_lo = q0 - WLOC; if (kb_lo < 0) kb_lo = 0; kb_lo &= ~31;
  const int kb_hi = (q0 + 15) & ~31;
  const int nblk  = ((kb_hi - kb_lo) >> 5) + 1;

  issue_async_kv(Kt, V, kb_lo, ldsK[0], ldsV[0], lane);

  for (int ib = 0; ib < nblk; ++ib) {
    const int kb = kb_lo + ib * 32;
    const int p  = ib & 1;
    const bool more = (ib + 1 < nblk);
    if (more)
      issue_async_kv(Kt, V, kb + 32, ldsK[p ^ 1], ldsV[p ^ 1], lane);
    // Current buffer done when <=16 async ops remain outstanding (in-order).
    if (more) asm volatile("s_wait_asynccnt 16" ::: "memory");
    else      asm volatile("s_wait_asynccnt 0"  ::: "memory");

    // K^T B-fragments from LDS: lane -> key col, elements -> d (table order).
    F16x16 kf[2][2];
#pragma unroll
    for (int tile = 0; tile < 2; ++tile)
#pragma unroll
      for (int c = 0; c < 2; ++c) {
#pragma unroll
        for (int s = 0; s < 16; ++s)
          kf[tile][c].h[s] = Kl[p][(c * 32 + hi0 + (s & 7) + ((s >> 3) << 4)) * 32 + tile * 16 + r];
      }

    v8f s0 = {}, s1 = {};
    s0 = wmma16(qa[0], kf[0][0], s0); s0 = wmma16(qa[1], kf[0][1], s0);
    s1 = wmma16(qa[0], kf[1][0], s1); s1 = wmma16(qa[1], kf[1][1], s1);

    // Scale + band/causal mask (finite sentinel; garbage rows self-correct).
    float p0[8], p1[8], rmx[8];
    const int jc = kb + r;
#pragma unroll
    for (int j = 0; j < 8; ++j) {
      const int i = q0 + j + hi0;
      const int jc1 = jc + 16;
      float u0 = ((jc  <= i) && (i - jc  <= WLOC)) ? s0[j] * SCORE_SCALE : NEGBIG;
      float u1 = ((jc1 <= i) && (i - jc1 <= WLOC)) ? s1[j] * SCORE_SCALE : NEGBIG;
      p0[j] = u0; p1[j] = u1;
      rmx[j] = fmaxf(u0, u1);
    }
    // Row max across the 16 lanes of this half.
#pragma unroll
    for (int msk = 1; msk < 16; msk <<= 1)
#pragma unroll
      for (int j = 0; j < 8; ++j)
        rmx[j] = fmaxf(rmx[j], __shfl_xor(rmx[j], msk, 32));

    float fac[8], rs[8];
#pragma unroll
    for (int j = 0; j < 8; ++j) {
      const float mn = fmaxf(mrun[j], rmx[j]);
      fac[j] = __expf(mrun[j] - mn);
      p0[j]  = __expf(p0[j] - mn);
      p1[j]  = __expf(p1[j] - mn);
      mrun[j] = mn;
      rs[j] = p0[j] + p1[j];
    }
#pragma unroll
    for (int msk = 1; msk < 16; msk <<= 1)
#pragma unroll
      for (int j = 0; j < 8; ++j)
        rs[j] += __shfl_xor(rs[j], msk, 32);
#pragma unroll
    for (int j = 0; j < 8; ++j) lrun[j] = lrun[j] * fac[j] + rs[j];
#pragma unroll
    for (int c = 0; c < 4; ++c)
#pragma unroll
      for (int j = 0; j < 8; ++j) o[c][j] *= fac[j];

    // P: C-layout f32 -> 16x32 f16 tile in LDS -> A-layout fragment.
#pragma unroll
    for (int j = 0; j < 8; ++j) {
      const int m = j + hi0;
      pl[m * 32 + r]      = (_Float16)p0[j];
      pl[m * 32 + r + 16] = (_Float16)p1[j];
    }
    __syncthreads();                      // single-wave block: orders DS ops
    F16x16 pa;
    {
      const _Float16* pr = pl + r * 32 + hi0;
      pa.q[0] = *(const uint4*)pr;
      pa.q[1] = *(const uint4*)(pr + 16);
    }
    __syncthreads();

    // PV: B = V tile from LDS (lane -> vd col, elements -> key, table order).
#pragma unroll
    for (int c = 0; c < 4; ++c) {
      F16x16 vf;
#pragma unroll
      for (int s = 0; s < 16; ++s)
        vf.h[s] = Vl[p][(hi0 + (s & 7) + ((s >> 3) << 4)) * 64 + c * 16 + r];
      o[c] = wmma16(pa, vf, o[c]);
    }
  }

  // Normalize + store attn[b][t][h*64 + vd] as f16.
#pragma unroll
  for (int j = 0; j < 8; ++j) {
    const float inv = 1.0f / lrun[j];
    const int t = q0 + j + hi0;
#pragma unroll
    for (int c = 0; c < 4; ++c)
      attn[((size_t)b * Tc + t) * DQc + h * HD + c * 16 + r] =
          (_Float16)(o[c][j] * inv);
  }
}

// ---------------------------------------------------------------------------
// Kernel 3: output projection  out = attn(f16) @ Wo + bo  (f32 out).
// 32x64 tile per wave, same structure as k_qkv.
// ---------------------------------------------------------------------------
__global__ __launch_bounds__(128) void k_out(
    const _Float16* __restrict__ attn, const float* __restrict__ Wo,
    const float* __restrict__ bo, float* __restrict__ out)
{
  const int lane = threadIdx.x;
  const int row0 = blockIdx.x * 32;
  const int n0   = (blockIdx.y * blockDim.y + threadIdx.y) * 64;
  const int r    = lane & 15;
  const int hi0  = (lane < 16) ? 0 : 8;

  v8f acc[2][4] = {};
  for (int kb = 0; kb < DQc; kb += 32) {
    F16x16 a[2];
#pragma unroll
    for (int rt = 0; rt < 2; ++rt) {
      const _Float16* ar = attn + (size_t)(row0 + rt * 16 + r) * DQc + kb + hi0;
      a[rt].q[0] = *(const uint4*)ar;
      a[rt].q[1] = *(const uint4*)(ar + 16);
    }
#pragma unroll
    for (int ct = 0; ct < 4; ++ct) {
      F16x16 bf;
      const float* wc = Wo + (size_t)kb * Dc + n0 + ct * 16 + r;
#pragma unroll
      for (int s = 0; s < 16; ++s)
        bf.h[s] = (_Float16)wc[(size_t)(hi0 + (s & 7) + ((s >> 3) << 4)) * Dc];
#pragma unroll
      for (int rt = 0; rt < 2; ++rt)
        acc[rt][ct] = wmma16(a[rt], bf, acc[rt][ct]);
    }
  }

#pragma unroll
  for (int ct = 0; ct < 4; ++ct) {
    const float bcol = bo[n0 + ct * 16 + r];
#pragma unroll
    for (int rt = 0; rt < 2; ++rt)
#pragma unroll
      for (int j = 0; j < 8; ++j)
        out[(size_t)(row0 + rt * 16 + j + hi0) * Dc + n0 + ct * 16 + r] =
            acc[rt][ct][j] + bcol;
  }
}

// ---------------------------------------------------------------------------
extern "C" void kernel_launch(void* const* d_in, const int* in_sizes, int n_in,
                              void* d_out, int out_size, void* d_ws, size_t ws_size,
                              hipStream_t stream) {
  const float* q_in = (const float*)d_in[0];
  const float* k_in = (const float*)d_in[1];
  const float* v_in = (const float*)d_in[2];
  const float* Wq   = (const float*)d_in[3];
  const float* bq   = (const float*)d_in[4];
  const float* Wk   = (const float*)d_in[5];
  const float* bk   = (const float*)d_in[6];
  const float* Wv   = (const float*)d_in[7];
  const float* bv   = (const float*)d_in[8];
  const float* Wo   = (const float*)d_in[9];
  const float* bo   = (const float*)d_in[10];
  float* out = (float*)d_out;

  // Workspace: 4 f16 buffers of B*H*T*64 == B*T*512 elements (8 MB each).
  const size_t NQ = (size_t)Bc * Hc * Tc * HD;
  _Float16* Qh   = (_Float16*)d_ws;
  _Float16* Kt   = Qh + NQ;
  _Float16* Vh   = Kt + NQ;
  _Float16* attn = Vh + NQ;

  // 1) QKV projections: 256 row-tiles x 8 col-tiles x {q,k,v}.
  dim3 gq((Bc * Tc) / 32, (DQc / 64) / 4, 3);
  k_qkv<<<gq, dim3(32, 4), 0, stream>>>(q_in, k_in, v_in, Wq, bq, Wk, bk,
                                        Wv, bv, Qh, Kt, Vh);

  // 2) Banded attention: one wave per (q-tile, head, batch).
  dim3 ga(Tc / 16, Hc, Bc);
  k_attn<<<ga, dim3(32, 1), 0, stream>>>(Qh, Kt, Vh, attn);

  // 3) Output projection.
  dim3 go((Bc * Tc) / 32, (Dc / 64) / 4);
  k_out<<<go, dim3(32, 4), 0, stream>>>(attn, Wo, bo, out);
}
